// DGNAgent_18975165513795
// MI455X (gfx1250) — matrix operand order repaired and checked by reading, not compiled, not run on hardware
//
#include <hip/hip_runtime.h>

typedef __attribute__((ext_vector_type(16))) _Float16 v16h;
typedef __attribute__((ext_vector_type(8)))  _Float16 v8h;
typedef __attribute__((ext_vector_type(4)))  _Float16 v4h;
typedef __attribute__((ext_vector_type(8)))  float    v8f;
typedef __attribute__((ext_vector_type(4)))  float    v4f;

#define WMMA_F16(a, b, c) \
  __builtin_amdgcn_wmma_f32_16x16x32_f16(false, (a), false, (b), (short)0, (c), false, false)

// ---------------------------------------------------------------------------
// Fragment loaders following CDNA5 16-bit WMMA VGPR layouts (ISA 7.12.2).
// A (16x32, MxK): lane<16 -> M=lane, halfs [K..K+7] then [K+16..K+23];
//                 lane>=16 -> M=lane-16, halfs [K+8..K+15] then [K+24..K+31].
// B (32x16, KxN): stored as W^T [N][K]; lane<16 -> N=lane, [K..K+7] then [K+16..K+23];
//                 lane>=16 -> N=lane-16, [K+8..K+15] then [K+24..K+31].
// D (16x16 f32):  col N = lane&15; elem j -> row M = (lane&16 ? 8 : 0)+j.
// ---------------------------------------------------------------------------
__device__ __forceinline__ v16h load_a_frag(const _Float16* base, int ldm,
                                            int mbase, int kbase, int lane) {
  int m  = mbase + (lane & 15);
  int k0 = kbase + ((lane & 16) ? 8 : 0);
  const _Float16* p = base + m * ldm + k0;
  v8h lo = *(const v8h*)(p);
  v8h hi = *(const v8h*)(p + 16);
  v16h r;
#pragma unroll
  for (int i = 0; i < 8; ++i) { r[i] = lo[i]; r[i + 8] = hi[i]; }
  return r;
}

__device__ __forceinline__ v16h load_b_frag(const _Float16* base, int ldk,
                                            int nbase, int kbase, int lane) {
  int n  = nbase + (lane & 15);
  int kk = kbase + ((lane & 16) ? 8 : 0);
  const _Float16* p = base + n * ldk + kk;
  v8h lo = *(const v8h*)(p);
  v8h hi = *(const v8h*)(p + 16);
  v16h r;
#pragma unroll
  for (int i = 0; i < 8; ++i) { r[i] = lo[i]; r[i + 8] = hi[i]; }
  return r;
}

// ---------------------------------------------------------------------------
// [32xK] (LDS, f16, row-major) x BT[128][K] (global f16, L2-resident) + bias
// -> [32][128]. Per wave: N-tiles {2w,2w+1}, both M-tiles. Optional ReLU;
// optional transposed store (dst[128][32]) so result is reusable as B operand.
// ---------------------------------------------------------------------------
template <int K, bool RELU, bool TRANS>
__device__ __forceinline__ void proj_gemm(const _Float16* __restrict__ A,
                                          const _Float16* __restrict__ BT,
                                          const float* __restrict__ bias,
                                          _Float16* __restrict__ dst,
                                          int wave, int lane) {
  v8f acc[2][2] = {};
#pragma unroll
  for (int kt = 0; kt < K / 32; ++kt) {
    int kb = kt * 32;
    v16h a0 = load_a_frag(A, K, 0,  kb, lane);
    v16h a1 = load_a_frag(A, K, 16, kb, lane);
    v16h b0 = load_b_frag(BT, K, wave * 32,      kb, lane);
    v16h b1 = load_b_frag(BT, K, wave * 32 + 16, kb, lane);
    acc[0][0] = WMMA_F16(a0, b0, acc[0][0]);
    acc[0][1] = WMMA_F16(a0, b1, acc[0][1]);
    acc[1][0] = WMMA_F16(a1, b0, acc[1][0]);
    acc[1][1] = WMMA_F16(a1, b1, acc[1][1]);
  }
  int ncol = lane & 15;
  int mo   = (lane & 16) ? 8 : 0;
#pragma unroll
  for (int mt = 0; mt < 2; ++mt) {
#pragma unroll
    for (int ntl = 0; ntl < 2; ++ntl) {
      int nb = wave * 32 + ntl * 16;
      int n  = nb + ncol;
      float bv = bias[n];
#pragma unroll
      for (int j = 0; j < 8; ++j) {
        int m = mt * 16 + mo + j;
        float v = acc[mt][ntl][j] + bv;
        if (RELU) v = fmaxf(v, 0.0f);
        if (TRANS) dst[n * 32 + m]  = (_Float16)v;
        else       dst[m * 128 + n] = (_Float16)v;
      }
    }
  }
}

// ---------------------------------------------------------------------------
// One attention block: v/q/k proj -> masked scores -> softmax -> att@v -> out
// proj. hbuf updated in place. vT holds V transposed so att@v uses it as B.
// ---------------------------------------------------------------------------
__device__ __forceinline__ void att_block(
    _Float16* hbuf, _Float16* qbuf, _Float16* kbuf, _Float16* vT,
    _Float16* obuf, float* att, _Float16* attf16,
    const _Float16* vwT, const _Float16* kwT,
    const _Float16* qwT, const _Float16* owT,
    const float* vb, const float* kb, const float* qb, const float* ob,
    const float* __restrict__ maskb, int wave, int lane, int tid) {
  proj_gemm<128, true, true >(hbuf, vwT, vb, vT,   wave, lane);
  proj_gemm<128, true, false>(hbuf, qwT, qb, qbuf, wave, lane);
  proj_gemm<128, true, false>(hbuf, kwT, kb, kbuf, wave, lane);
  __syncthreads();

  // scores[32][32] = q @ k^T, one 16x16 tile per wave, K=128
  {
    int mt = wave >> 1, nt = wave & 1;
    v8f acc = {};
#pragma unroll
    for (int kt = 0; kt < 4; ++kt) {
      v16h a = load_a_frag(qbuf, 128, mt * 16, kt * 32, lane);
      v16h b = load_b_frag(kbuf, 128, nt * 16, kt * 32, lane);
      acc = WMMA_F16(a, b, acc);
    }
    int n  = nt * 16 + (lane & 15);
    int mo = (lane & 16) ? 8 : 0;
#pragma unroll
    for (int j = 0; j < 8; ++j) {
      int m = mt * 16 + mo + j;
      // mask is streamed once -> non-temporal load, keep weights cached
      float mk = __builtin_nontemporal_load(maskb + m * 32 + n);
      att[m * 32 + n] = acc[j] * mk - 9e15f * (1.0f - mk);
    }
  }
  __syncthreads();

  // softmax over j, one row per lane of wave 0 (wave-uniform branch, no WMMA)
  if (tid < 32) {
    int r = tid;
    float mx = -3.4e38f;
    for (int j = 0; j < 32; ++j) mx = fmaxf(mx, att[r * 32 + j]);
    float s = 0.0f;
    for (int j = 0; j < 32; ++j) s += __expf(att[r * 32 + j] - mx);
    float inv = 1.0f / s;
    for (int j = 0; j < 32; ++j)
      attf16[r * 32 + j] = (_Float16)(__expf(att[r * 32 + j] - mx) * inv);
  }
  __syncthreads();

  // out[32][128] = att[32][32] @ v[32][128]; single K=32 step per tile
  {
    int ncol = lane & 15;
    int mo   = (lane & 16) ? 8 : 0;
#pragma unroll
    for (int mt = 0; mt < 2; ++mt) {
#pragma unroll
      for (int ntl = 0; ntl < 2; ++ntl) {
        int nb = wave * 32 + ntl * 16;
        v8f acc = {};
        v16h a = load_a_frag(attf16, 32, mt * 16, 0, lane);
        v16h b = load_b_frag(vT, 32, nb, 0, lane);
        acc = WMMA_F16(a, b, acc);
#pragma unroll
        for (int j = 0; j < 8; ++j)
          obuf[(mt * 16 + mo + j) * 128 + nb + ncol] = (_Float16)acc[j];
      }
    }
  }
  __syncthreads();

  proj_gemm<128, true, false>(obuf, owT, ob, hbuf, wave, lane);
  __syncthreads();
}

// ---------------------------------------------------------------------------
// Main kernel: one workgroup (4 wave32) per batch element.
// ---------------------------------------------------------------------------
__global__ __launch_bounds__(128) void dgn_kernel(
    const float* __restrict__ x, const float* __restrict__ mask,
    const _Float16* __restrict__ wsh,
    const float* __restrict__ enc_b,
    const float* __restrict__ a1_vb, const float* __restrict__ a1_kb,
    const float* __restrict__ a1_qb, const float* __restrict__ a1_ob,
    const float* __restrict__ a2_vb, const float* __restrict__ a2_kb,
    const float* __restrict__ a2_qb, const float* __restrict__ a2_ob,
    const float* __restrict__ q_b, float* __restrict__ out) {
  __shared__ __align__(32) _Float16 xs[32 * 256];     // 16 KB
  __shared__ __align__(32) _Float16 hbuf[32 * 128];   // 8 KB
  __shared__ __align__(32) _Float16 qbuf[32 * 128];   // 8 KB
  __shared__ __align__(32) _Float16 kbuf[32 * 128];   // 8 KB
  __shared__ __align__(32) _Float16 vT[128 * 32];     // 8 KB (V transposed)
  __shared__ __align__(32) _Float16 obuf[32 * 128];   // 8 KB
  __shared__ __align__(32) float    att[32 * 32];     // 4 KB
  __shared__ __align__(32) _Float16 attf16[32 * 32];  // 2 KB

  int b = blockIdx.x;
  int tid = threadIdx.x;
  int wave = tid >> 5, lane = tid & 31;

  // stage x[b] into LDS as f16: vectorized b128 non-temporal loads (x is
  // streamed exactly once; don't let 128MB of stream evict the weights)
  {
    const v4f* xb4 = (const v4f*)(x + (size_t)b * 32 * 256);
#pragma unroll
    for (int i = tid; i < 32 * 256 / 4; i += 128) {
      v4f f = __builtin_nontemporal_load(xb4 + i);
      v4h h;
      h[0] = (_Float16)f[0]; h[1] = (_Float16)f[1];
      h[2] = (_Float16)f[2]; h[3] = (_Float16)f[3];
      *(v4h*)(xs + i * 4) = h;
    }
  }
  __syncthreads();

  const _Float16* enc_wT = wsh;                        // [128][256]
  const _Float16* wT     = wsh + 32768;                // 8 x [128][128]
  const _Float16* q_wT   = wsh + 32768 + 8 * 16384;    // [32][128]

  proj_gemm<256, true, false>(xs, enc_wT, enc_b, hbuf, wave, lane);
  __syncthreads();

  const float* maskb = mask + (size_t)b * 32 * 32;
  att_block(hbuf, qbuf, kbuf, vT, obuf, att, attf16,
            wT + 0 * 16384, wT + 1 * 16384, wT + 2 * 16384, wT + 3 * 16384,
            a1_vb, a1_kb, a1_qb, a1_ob, maskb, wave, lane, tid);
  att_block(hbuf, qbuf, kbuf, vT, obuf, att, attf16,
            wT + 4 * 16384, wT + 5 * 16384, wT + 6 * 16384, wT + 7 * 16384,
            a2_vb, a2_kb, a2_qb, a2_ob, maskb, wave, lane, tid);

  // final head: qout[32][32] = hbuf @ q_w + q_b, one tile per wave
  {
    int mt = wave >> 1, nt = wave & 1;
    v8f acc = {};
#pragma unroll
    for (int kt = 0; kt < 4; ++kt) {
      v16h a = load_a_frag(hbuf, 128, mt * 16, kt * 32, lane);
      v16h bf = load_b_frag(q_wT, 128, nt * 16, kt * 32, lane);
      acc = WMMA_F16(a, bf, acc);
    }
    int n  = nt * 16 + (lane & 15);
    int mo = (lane & 16) ? 8 : 0;
    float* outb = out + (size_t)b * 32 * 32;
#pragma unroll
    for (int j = 0; j < 8; ++j) {
      float v = acc[j] + q_b[n];
      // output is written once -> non-temporal store
      __builtin_nontemporal_store(v, outb + (mt * 16 + mo + j) * 32 + n);
    }
  }
}

// ---------------------------------------------------------------------------
// Weight prep: transpose [R][C] f32 -> [C][R] f16 (so K is contiguous per
// output column, matching the WMMA B-fragment loader).
// ---------------------------------------------------------------------------
__global__ void transpose_f32_to_f16(const float* __restrict__ src,
                                     _Float16* __restrict__ dst, int R, int C) {
  int idx = blockIdx.x * blockDim.x + threadIdx.x;
  if (idx < R * C) {
    int r = idx / C, c = idx % C;
    dst[c * R + r] = (_Float16)src[idx];
  }
}

extern "C" void kernel_launch(void* const* d_in, const int* in_sizes, int n_in,
                              void* d_out, int out_size, void* d_ws, size_t ws_size,
                              hipStream_t stream) {
  (void)in_sizes; (void)n_in; (void)out_size; (void)ws_size;
  _Float16* wsh = (_Float16*)d_ws;

  auto T = [&](const void* src, _Float16* dst, int R, int C) {
    int n = R * C;
    transpose_f32_to_f16<<<(n + 255) / 256, 256, 0, stream>>>(
        (const float*)src, dst, R, C);
  };

  // inputs: 0 x, 1 mask, 2 enc_w, 3 enc_b, 4 a1_vw, 5 a1_vb, 6 a1_kw, 7 a1_kb,
  // 8 a1_qw, 9 a1_qb, 10 a1_ow, 11 a1_ob, 12..19 a2_*, 20 q_w, 21 q_b
  T(d_in[2],  wsh,                     256, 128); // enc_wT [128][256]
  T(d_in[4],  wsh + 32768 + 0 * 16384, 128, 128); // a1_vw
  T(d_in[6],  wsh + 32768 + 1 * 16384, 128, 128); // a1_kw
  T(d_in[8],  wsh + 32768 + 2 * 16384, 128, 128); // a1_qw
  T(d_in[10], wsh + 32768 + 3 * 16384, 128, 128); // a1_ow
  T(d_in[12], wsh + 32768 + 4 * 16384, 128, 128); // a2_vw
  T(d_in[14], wsh + 32768 + 5 * 16384, 128, 128); // a2_kw
  T(d_in[16], wsh + 32768 + 6 * 16384, 128, 128); // a2_qw
  T(d_in[18], wsh + 32768 + 7 * 16384, 128, 128); // a2_ow
  T(d_in[20], wsh + 32768 + 8 * 16384, 128, 32);  // q_wT [32][128]

  dgn_kernel<<<4096, 128, 0, stream>>>(
      (const float*)d_in[0], (const float*)d_in[1], wsh,
      (const float*)d_in[3],
      (const float*)d_in[5],  (const float*)d_in[7],
      (const float*)d_in[9],  (const float*)d_in[11],
      (const float*)d_in[13], (const float*)d_in[15],
      (const float*)d_in[17], (const float*)d_in[19],
      (const float*)d_in[21], (float*)d_out);
}
